// MTMamba_62792421868260
// MI455X (gfx1250) — compile-verified, matrix operand order checked by reading, not run on hardware
//
#include <hip/hip_runtime.h>
#include <hip/hip_bf16.h>
#include <math.h>

#define D_INPUT  900
#define D_MODEL  900
#define D_STATE  16
#define D_CONV   4
#define D_INNER  1800
#define DT_RANK  57
#define B_SZ     2
#define SEQ_L    900
#define BLROWS   (B_SZ * SEQ_L)
#define XD_DIM   (DT_RANK + 2 * D_STATE)   // 89

typedef __bf16 bf16;
typedef __attribute__((ext_vector_type(16))) __bf16 v16bf;
typedef __attribute__((ext_vector_type(8)))  float  v8f;

__device__ __forceinline__ float sigf(float x) { return 1.0f / (1.0f + __expf(-x)); }

// ---------------- pack f32 [M,K] (row stride ld) into WMMA fragment tiles ----------------
// layout: out[((mt*ktiles + kt)*32 + lane)*16 + j]  (bf16, zero padded)
// lane maps row: m = mt*16 + (lane&15); half = lane>>4
// element j: v = j>>1; k = kt*32 + ((v&4)<<2) + (half<<3) + ((v&3)<<1) + (j&1)
__global__ void k_pack(const float* __restrict__ in, bf16* __restrict__ out,
                       int M, int K, int ld) {
    int ktiles = (K + 31) / 32;
    size_t total = (size_t)((M + 15) / 16) * ktiles * 512;
    size_t i = (size_t)blockIdx.x * blockDim.x + threadIdx.x;
    if (i >= total) return;
    int j    = (int)(i & 15);
    int lane = (int)((i >> 4) & 31);
    size_t t = i >> 9;                       // mt*ktiles + kt
    int kt = (int)(t % ktiles);
    int mt = (int)(t / ktiles);
    int m = mt * 16 + (lane & 15);
    int v = j >> 1;
    int k = kt * 32 + ((v & 4) << 2) + ((lane >> 4) << 3) + ((v & 3) << 1) + (j & 1);
    float val = (m < M && k < K) ? in[(size_t)m * ld + k] : 0.f;
    out[i] = (bf16)val;
}

// A = -exp(A_log)
__global__ void k_aneg(const float* __restrict__ alog, float* __restrict__ out, int n) {
    int i = blockIdx.x * blockDim.x + threadIdx.x;
    if (i < n) out[i] = -expf(alog[i]);
}

// x = x*sqrt(D_MODEL) + pe[l, c]
__global__ void k_scale_pe(float* __restrict__ x, const float* __restrict__ pe, int n) {
    int i = blockIdx.x * blockDim.x + threadIdx.x;
    if (i < n) {
        int c = i % D_MODEL;
        int r = i / D_MODEL;
        int l = r % SEQ_L;
        x[i] = x[i] * 30.0f + pe[(size_t)l * D_MODEL + c];   // sqrt(900)=30
    }
}

// ---------------- packed bf16 WMMA GEMM: C[M,N] = A @ B^T (+bias[n]) (+addend[m,n]) ----------------
// Ap: packed [mtiles][ktiles][32][16], Bp: packed [ntiles][ktiles][32][16]
// one wave computes a 16x64 strip (4 N-tiles); 256-thread block = 8 waves = 128 rows
__global__ void k_gemm(const bf16* __restrict__ Ap, const bf16* __restrict__ Bp,
                       const float* __restrict__ bias, const float* __restrict__ addend,
                       float* __restrict__ C, int M, int N, int K) {
    int ktiles = (K + 31) / 32;
    int ntiles = (N + 15) / 16;
    int lane = threadIdx.x & 31;
    int wave = threadIdx.x >> 5;
    int mt  = blockIdx.y * 8 + wave;
    if (mt * 16 >= M) return;                      // uniform per wave
    int nt0 = blockIdx.x * 4;

    const bf16* ap = Ap + ((size_t)mt * ktiles * 32 + lane) * 16;
    const bf16* bp0; const bf16* bp1; const bf16* bp2; const bf16* bp3;
    {
        int n0 = nt0 + 0; if (n0 >= ntiles) n0 = ntiles - 1;
        int n1 = nt0 + 1; if (n1 >= ntiles) n1 = ntiles - 1;
        int n2 = nt0 + 2; if (n2 >= ntiles) n2 = ntiles - 1;
        int n3 = nt0 + 3; if (n3 >= ntiles) n3 = ntiles - 1;
        bp0 = Bp + ((size_t)n0 * ktiles * 32 + lane) * 16;
        bp1 = Bp + ((size_t)n1 * ktiles * 32 + lane) * 16;
        bp2 = Bp + ((size_t)n2 * ktiles * 32 + lane) * 16;
        bp3 = Bp + ((size_t)n3 * ktiles * 32 + lane) * 16;
    }

    v8f acc0 = {0.f,0.f,0.f,0.f,0.f,0.f,0.f,0.f};
    v8f acc1 = acc0, acc2 = acc0, acc3 = acc0;

    for (int kt = 0; kt < ktiles; ++kt) {
        __builtin_prefetch(ap + 512, 0, 1);        // gfx1250 global_prefetch_b8
        v16bf a  = *(const v16bf*)ap;  ap  += 512;
        v16bf b0 = *(const v16bf*)bp0; bp0 += 512;
        v16bf b1 = *(const v16bf*)bp1; bp1 += 512;
        v16bf b2 = *(const v16bf*)bp2; bp2 += 512;
        v16bf b3 = *(const v16bf*)bp3; bp3 += 512;
        acc0 = __builtin_amdgcn_wmma_f32_16x16x32_bf16(false, a, false, b0, (short)0, acc0, false, false);
        acc1 = __builtin_amdgcn_wmma_f32_16x16x32_bf16(false, a, false, b1, (short)0, acc1, false, false);
        acc2 = __builtin_amdgcn_wmma_f32_16x16x32_bf16(false, a, false, b2, (short)0, acc2, false, false);
        acc3 = __builtin_amdgcn_wmma_f32_16x16x32_bf16(false, a, false, b3, (short)0, acc3, false, false);
    }

    int r = lane & 15, half = lane >> 4;
#pragma unroll
    for (int j = 0; j < 4; ++j) {
        v8f acc = (j == 0) ? acc0 : (j == 1) ? acc1 : (j == 2) ? acc2 : acc3;
        int n = (nt0 + j) * 16 + r;
        if (n >= N) continue;
        float bn = bias ? bias[n] : 0.f;
#pragma unroll
        for (int i = 0; i < 8; ++i) {
            int m = mt * 16 + (half << 3) + i;     // C layout: VGPR i -> M=i (+8 upper half)
            if (m < M) {
                float v = acc[i] + bn;
                if (addend) v += addend[(size_t)m * N + n];
                C[(size_t)m * N + n] = v;
            }
        }
    }
}

// ---------------- RMSNorm: one block per row, f32 out ----------------
__global__ void k_rmsnorm(const float* __restrict__ x, const float* __restrict__ w,
                          float* __restrict__ out_f, int D) {
    int row = blockIdx.x;
    const float* xr = x + (size_t)row * D;
    __shared__ float red[8];
    float s = 0.f;
    for (int i = threadIdx.x; i < D; i += blockDim.x) { float v = xr[i]; s += v * v; }
    for (int off = 16; off > 0; off >>= 1) s += __shfl_down(s, off, 32);
    if ((threadIdx.x & 31) == 0) red[threadIdx.x >> 5] = s;
    __syncthreads();
    if (threadIdx.x == 0) {
        float t = 0.f;
        for (int i = 0; i < (int)(blockDim.x >> 5); ++i) t += red[i];
        red[0] = rsqrtf(t / (float)D + 1e-5f);
    }
    __syncthreads();
    float rs = red[0];
    for (int i = threadIdx.x; i < D; i += blockDim.x)
        out_f[(size_t)row * D + i] = xr[i] * rs * w[i];
}

// ---------------- causal depthwise conv (k=4) + SiLU, fwd + channel-reversed ----------------
__global__ void k_conv_silu(const float* __restrict__ xr,
                            const float* __restrict__ cw, const float* __restrict__ cb,
                            float* __restrict__ u_f, float* __restrict__ u_b, int n) {
    int i = blockIdx.x * blockDim.x + threadIdx.x;
    if (i >= n) return;
    int d = i % D_INNER;
    int r = i / D_INNER;
    int l = r % SEQ_L;
    int b = r / SEQ_L;
    const float* base = xr + (size_t)b * SEQ_L * (2 * D_INNER);
    float accf = cb[d], accb = cb[d];
#pragma unroll
    for (int j = 0; j < D_CONV; ++j) {
        int t = l - (D_CONV - 1) + j;
        if (t >= 0) {
            float w = cw[d * D_CONV + j];
            accf += w * base[(size_t)t * (2 * D_INNER) + d];
            accb += w * base[(size_t)t * (2 * D_INNER) + (D_INNER - 1 - d)];
        }
    }
    u_f[i] = accf * sigf(accf);
    u_b[i] = accb * sigf(accb);
}

// ---------------- selective scan: sequential over L, parallel over channels ----------------
__global__ void k_scan(const float* __restrict__ u, const float* __restrict__ delta_pre,
                       const float* __restrict__ xd, const float* __restrict__ Aneg,
                       const float* __restrict__ Dv, float* __restrict__ y) {
    int d = blockIdx.x * blockDim.x + threadIdx.x;
    int b = blockIdx.y;
    bool act = (d < D_INNER);
    __shared__ float sB[2][D_STATE];
    __shared__ float sC[2][D_STATE];
    float Areg[D_STATE];
    float h[D_STATE];
#pragma unroll
    for (int nn = 0; nn < D_STATE; ++nn) {
        Areg[nn] = act ? Aneg[(size_t)d * D_STATE + nn] : 0.f;
        h[nn] = 0.f;
    }
    float Dd = act ? Dv[d] : 0.f;

    for (int l = 0; l < SEQ_L; ++l) {
        int buf = l & 1;
        if (threadIdx.x < 32) {
            const float* row = xd + (size_t)(b * SEQ_L + l) * XD_DIM + DT_RANK;
            if (threadIdx.x < D_STATE) sB[buf][threadIdx.x] = row[threadIdx.x];
            else                       sC[buf][threadIdx.x - D_STATE] = row[threadIdx.x];
        }
        __syncthreads();
        if (act) {
            size_t off = (size_t)(b * SEQ_L + l) * D_INNER + d;
            float dl = delta_pre[off];
            dl = (dl > 20.f) ? dl : log1pf(__expf(dl));   // softplus
            float uu = u[off];
            float accy = 0.f;
#pragma unroll
            for (int nn = 0; nn < D_STATE; ++nn) {
                h[nn] = __expf(dl * Areg[nn]) * h[nn] + dl * sB[buf][nn] * uu;
                accy += h[nn] * sC[buf][nn];
            }
            y[off] = accy + uu * Dd;
        }
    }
}

// ---------------- gated combine: z = (y_f + rev(y_b)) * silu(res), f32 out ----------------
__global__ void k_combine(const float* __restrict__ y_f, const float* __restrict__ y_b,
                          const float* __restrict__ xr, float* __restrict__ z, int n) {
    int i = blockIdx.x * blockDim.x + threadIdx.x;
    if (i >= n) return;
    int d = i % D_INNER;
    int r = i / D_INNER;
    float res = xr[(size_t)r * (2 * D_INNER) + D_INNER + d];
    float g = res * sigf(res);
    z[i] = (y_f[i] + y_b[(size_t)r * D_INNER + (D_INNER - 1 - d)]) * g;
}

// =====================================================================
extern "C" void kernel_launch(void* const* d_in, const int* in_sizes, int n_in,
                              void* d_out, int out_size, void* d_ws, size_t ws_size,
                              hipStream_t stream) {
    (void)in_sizes; (void)n_in; (void)out_size; (void)ws_size;
    const float* inp     = (const float*)d_in[0];
    const float* W_emb   = (const float*)d_in[1];
    const float* b_emb   = (const float*)d_in[2];
    const float* norm_w  = (const float*)d_in[3];
    const float* W_in    = (const float*)d_in[4];
    const float* conv_w  = (const float*)d_in[5];
    const float* conv_b  = (const float*)d_in[6];
    const float* W_xp    = (const float*)d_in[7];
    const float* W_dt    = (const float*)d_in[8];
    const float* b_dt    = (const float*)d_in[9];
    const float* A_log   = (const float*)d_in[10];
    const float* Dv      = (const float*)d_in[11];
    const float* W_out   = (const float*)d_in[12];
    const float* normf_w = (const float*)d_in[13];
    const float* pe      = (const float*)d_in[14];

    // ---- workspace bump allocator ----
    char* p = (char*)d_ws;
    auto alloc = [&](size_t bytes) -> void* {
        void* r = (void*)p;
        p += (bytes + 255) & ~(size_t)255;
        return r;
    };
    auto ptile = [](int M, int K) -> size_t {   // packed element count
        return (size_t)((M + 15) / 16) * ((K + 31) / 32) * 512;
    };

    // packed bf16 operands
    bf16* inp_p  = (bf16*)alloc(ptile(BLROWS, D_INPUT) * 2);
    bf16* Wemb_p = (bf16*)alloc(ptile(D_MODEL, D_INPUT) * 2);
    bf16* xn_p   = (bf16*)alloc(ptile(BLROWS, D_MODEL) * 2);
    bf16* Win_p  = (bf16*)alloc(ptile(2 * D_INNER, D_MODEL) * 2);
    bf16* uf_p   = (bf16*)alloc(ptile(BLROWS, D_INNER) * 2);
    bf16* ub_p   = (bf16*)alloc(ptile(BLROWS, D_INNER) * 2);
    bf16* Wxp_p  = (bf16*)alloc(ptile(XD_DIM, D_INNER) * 2);
    bf16* dtf_p  = (bf16*)alloc(ptile(BLROWS, DT_RANK) * 2);
    bf16* dtb_p  = (bf16*)alloc(ptile(BLROWS, DT_RANK) * 2);
    bf16* Wdt_p  = (bf16*)alloc(ptile(D_INNER, DT_RANK) * 2);
    bf16* z_p    = (bf16*)alloc(ptile(BLROWS, D_INNER) * 2);
    bf16* Wout_p = (bf16*)alloc(ptile(D_MODEL, D_INNER) * 2);
    // f32 intermediates
    float* Aneg    = (float*)alloc((size_t)D_INNER * D_STATE * 4);
    float* x       = (float*)alloc((size_t)BLROWS * D_MODEL * 4);
    float* xn      = (float*)alloc((size_t)BLROWS * D_MODEL * 4);
    float* xr      = (float*)alloc((size_t)BLROWS * 2 * D_INNER * 4);
    float* u_f     = (float*)alloc((size_t)BLROWS * D_INNER * 4);
    float* u_b     = (float*)alloc((size_t)BLROWS * D_INNER * 4);
    float* xd_f    = (float*)alloc((size_t)BLROWS * XD_DIM * 4);
    float* xd_b    = (float*)alloc((size_t)BLROWS * XD_DIM * 4);
    float* delta_f = (float*)alloc((size_t)BLROWS * D_INNER * 4);
    float* delta_b = (float*)alloc((size_t)BLROWS * D_INNER * 4);
    float* y_f     = (float*)alloc((size_t)BLROWS * D_INNER * 4);
    float* y_b     = (float*)alloc((size_t)BLROWS * D_INNER * 4);
    float* z       = (float*)alloc((size_t)BLROWS * D_INNER * 4);
    float* y_pre   = (float*)alloc((size_t)BLROWS * D_MODEL * 4);

    auto ewg = [](size_t n) { return dim3((unsigned)((n + 255) / 256)); };
    auto pack = [&](const float* in, bf16* out, int M, int K, int ld) {
        size_t total = (size_t)((M + 15) / 16) * ((K + 31) / 32) * 512;
        k_pack<<<ewg(total), 256, 0, stream>>>(in, out, M, K, ld);
    };
    auto gemm = [&](const bf16* A, const bf16* B, const float* bias, const float* add,
                    float* C, int M, int N, int K) {
        int mtiles = (M + 15) / 16, ntiles = (N + 15) / 16;
        dim3 g((unsigned)((ntiles + 3) / 4), (unsigned)((mtiles + 7) / 8));
        k_gemm<<<g, 256, 0, stream>>>(A, B, bias, add, C, M, N, K);
    };

    // weight packs (operand B of each GEMM) + misc precompute
    pack(W_emb, Wemb_p, D_MODEL, D_INPUT, D_INPUT);
    pack(W_in,  Win_p,  2 * D_INNER, D_MODEL, D_MODEL);
    pack(W_xp,  Wxp_p,  XD_DIM, D_INNER, D_INNER);
    pack(W_dt,  Wdt_p,  D_INNER, DT_RANK, DT_RANK);
    pack(W_out, Wout_p, D_MODEL, D_INNER, D_INNER);
    k_aneg<<<ewg(D_INNER * D_STATE), 256, 0, stream>>>(A_log, Aneg, D_INNER * D_STATE);

    // 1) x = inp @ W_emb^T + b_emb ; scale + positional encoding
    pack(inp, inp_p, BLROWS, D_INPUT, D_INPUT);
    gemm(inp_p, Wemb_p, b_emb, nullptr, x, BLROWS, D_MODEL, D_INPUT);
    k_scale_pe<<<ewg((size_t)BLROWS * D_MODEL), 256, 0, stream>>>(x, pe, BLROWS * D_MODEL);

    // 2) rmsnorm, pack
    k_rmsnorm<<<BLROWS, 256, 0, stream>>>(x, norm_w, xn, D_MODEL);
    pack(xn, xn_p, BLROWS, D_MODEL, D_MODEL);

    // 3) in-projection: xr = xn @ W_in^T  (900 -> 3600)
    gemm(xn_p, Win_p, nullptr, nullptr, xr, BLROWS, 2 * D_INNER, D_MODEL);

    // 4) causal depthwise conv + SiLU (fwd + channel-reversed), pack both
    k_conv_silu<<<ewg((size_t)BLROWS * D_INNER), 256, 0, stream>>>(xr, conv_w, conv_b,
                                                                   u_f, u_b, BLROWS * D_INNER);
    pack(u_f, uf_p, BLROWS, D_INNER, D_INNER);
    pack(u_b, ub_p, BLROWS, D_INNER, D_INNER);

    // 5) x-projection: xd = u @ W_xp^T  (1800 -> 89)
    gemm(uf_p, Wxp_p, nullptr, nullptr, xd_f, BLROWS, XD_DIM, D_INNER);
    gemm(ub_p, Wxp_p, nullptr, nullptr, xd_b, BLROWS, XD_DIM, D_INNER);

    // 6) delta_pre = xd[:, :57] @ W_dt^T + b_dt  (softplus fused into scan)
    pack(xd_f, dtf_p, BLROWS, DT_RANK, XD_DIM);   // strided slice via ld=89
    pack(xd_b, dtb_p, BLROWS, DT_RANK, XD_DIM);
    gemm(dtf_p, Wdt_p, b_dt, nullptr, delta_f, BLROWS, D_INNER, DT_RANK);
    gemm(dtb_p, Wdt_p, b_dt, nullptr, delta_b, BLROWS, D_INNER, DT_RANK);

    // 7) selective scans (sequential over L, parallel over channels)
    dim3 sg((unsigned)((D_INNER + 255) / 256), (unsigned)B_SZ);
    k_scan<<<sg, 256, 0, stream>>>(u_f, delta_f, xd_f, Aneg, Dv, y_f);
    k_scan<<<sg, 256, 0, stream>>>(u_b, delta_b, xd_b, Aneg, Dv, y_b);

    // 8) gated combine, pack
    k_combine<<<ewg((size_t)BLROWS * D_INNER), 256, 0, stream>>>(y_f, y_b, xr, z, BLROWS * D_INNER);
    pack(z, z_p, BLROWS, D_INNER, D_INNER);

    // 9) out-projection + residual: y_pre = z @ W_out^T + x
    gemm(z_p, Wout_p, nullptr, x, y_pre, BLROWS, D_MODEL, D_INNER);

    // 10) final rmsnorm straight into d_out (f32 logits, contiguous)
    k_rmsnorm<<<BLROWS, 256, 0, stream>>>(y_pre, normf_w, (float*)d_out, D_MODEL);
}
